// conv_59700045414486
// MI455X (gfx1250) — matrix operand order repaired and checked by reading, not compiled, not run on hardware
//
#include <hip/hip_runtime.h>

// CDNA5 / gfx1250 lattice-gather GEMM:
//   y[b,n,o] = sum_{k,c} x[b, kernel2[n,k], c] * W[k,c,o] + bias[o]
// 8 sites per workgroup. Per site: M=64(batch) x N=64(OUT) x K=256 GEMM via
// v_wmma_f32_16x16x32_f16. Gathered x rows are staged raw (f32) into LDS with
// GLOBAL_LOAD_ASYNC_TO_LDS_B128 double-buffering (copy of site s+1 overlaps
// WMMA of site s), converted to f16 inside the fragment load. Neighbor index k
// is kept wave-uniform in the DMA loop so the issue path is branch-free.
// Output uses non-temporal stores so x stays resident in the 192MB L2.

typedef _Float16 half_t;
typedef __attribute__((ext_vector_type(16))) _Float16 v16h;
typedef __attribute__((ext_vector_type(8)))  _Float16 v8h;
typedef __attribute__((ext_vector_type(2)))  _Float16 v2h;
typedef __attribute__((ext_vector_type(8)))  float    v8f;
typedef __attribute__((ext_vector_type(4)))  float    f32x4;

#define B_    64
#define N_    4096
#define C_    64
#define O_    64
#define K2_   4
#define SITES 8                       // lattice sites per workgroup

#define RSW_  264                     // wS row stride (halfs): 256 + 8 pad
#define RSX_  260                     // raw x row stride (floats): 256 + 4 pad

#define WS_BYTES   (64 * RSW_ * 2)            // 33,792
#define RAW_BYTES  (64 * RSX_ * 4)            // 66,560
#define RAW0_OFF   ((unsigned)WS_BYTES)
#define RAW1_OFF   ((unsigned)(WS_BYTES + RAW_BYTES))
#define SMEM_BYTES (WS_BYTES + 2 * RAW_BYTES) // 166,912 of 320KB WGP LDS

extern "C" __global__ __launch_bounds__(256)
void lattice_conv_wmma_async(const float* __restrict__ x,
                             const float* __restrict__ W,
                             const float* __restrict__ bias,
                             const int*   __restrict__ kernel2,
                             float* __restrict__ out)
{
    extern __shared__ char smem[];
    half_t* __restrict__ wS = (half_t*)smem;   // [64][RSW_] transposed W (f16)

    const int tid  = threadIdx.x;
    const int lane = tid & 31;                 // wave32
    const int wv   = tid >> 5;                 // 8 waves
    const int n0   = blockIdx.x * SITES;

    const unsigned long long xBase = (unsigned long long)x;

    // Per-lane address components for the DMA chunks, reused for all sites.
    // Sub-pass 'sub' covers chunks ch = tid + sub*256 (0..1023): row b = ch>>4,
    // 16B chunk j = ch&15 -> 16 consecutive lanes read 256B contiguous.
    unsigned base_g[4], base_l[4];
#pragma unroll
    for (int sub = 0; sub < 4; ++sub) {
        const int ch = tid + sub * 256;
        const int b  = ch >> 4;
        const int j  = ch & 15;
        base_g[sub] = (unsigned)(b * (N_ * C_ * 4) + j * 16);
        base_l[sub] = (unsigned)(b * (RSX_ * 4) + j * 16);
    }

    // ---- issue async DMA of one site's gathered rows into a raw LDS buffer --
    // k is uniform per unrolled iteration: gather index stays in SGPRs, the
    // issue path is branch-free (2 v_adds + 1 DMA per 16B chunk).
    auto issue_site = [&](int s, unsigned rawOff) {
        const int base = (n0 + s) * K2_;
        int idx[4];
#pragma unroll
        for (int k = 0; k < 4; ++k) idx[k] = kernel2[base + k];
#pragma unroll
        for (int k = 0; k < 4; ++k) {
            const unsigned gk = (unsigned)idx[k] * (C_ * 4);      // scalar
            const unsigned lk = rawOff + (unsigned)(k * 64 * 4);  // scalar
#pragma unroll
            for (int sub = 0; sub < 4; ++sub) {
                const unsigned gOff = base_g[sub] + gk;
                const unsigned lOff = base_l[sub] + lk;
                asm volatile("global_load_async_to_lds_b128 %0, %1, %2"
                             :: "v"(lOff), "v"(gOff), "s"(xBase)
                             : "memory");
            }
        }
    };

    // Kick off site 0's DMA first so it overlaps W staging.
    issue_site(0, RAW0_OFF);

    // ---- stage W transposed + cvt f32->f16: wS[o][kk] (once per block) ------
#pragma unroll
    for (int it = 0; it < 32; ++it) {
        const int e  = tid + it * 256;
        const int o  = e & 63;
        const int kp = e >> 6;                 // kk pair
        const float w0 = W[(2 * kp    ) * O_ + o];
        const float w1 = W[(2 * kp + 1) * O_ + o];
        v2h h; h[0] = (half_t)w0; h[1] = (half_t)w1;
        *(v2h*)(wS + o * RSW_ + 2 * kp) = h;
    }

    asm volatile("s_wait_asynccnt 0x0" ::: "memory");
    __syncthreads();

    // ---- per-wave tile assignment (two adjacent 16x16 tiles, shared M) ------
    const int tm   = wv >> 1;                  // M tile 0..3
    const int tn0  = (wv & 1) * 2;             // first N tile (0 or 2)
    const int l16  = lane & 15;
    const int hi16 = lane >> 4;

    const half_t* wRow0 = wS + (tn0 * 16 + l16) * RSW_;
    const half_t* wRow1 = wRow0 + 16 * RSW_;
    const float   bb0   = bias[tn0 * 16      + l16];
    const float   bb1   = bias[tn0 * 16 + 16 + l16];

#pragma unroll 1
    for (int s = 0; s < SITES; ++s) {
        // DMA next site while this one computes.
        if (s + 1 < SITES)
            issue_site(s + 1, ((s + 1) & 1) ? RAW1_OFF : RAW0_OFF);

        const float* rawA  = (const float*)(smem + ((s & 1) ? RAW1_OFF : RAW0_OFF));
        const float* aRowF = rawA + (tm * 16 + l16) * RSX_;

        v8f c0 = {};
        v8f c1 = {};

#pragma unroll
        for (int ks = 0; ks < 8; ++ks) {
            const int ko = ks * 32;
            union { v16h v; v8h h[2]; } a, b0, b1;

            // A fragment from raw f32 rows, converted in-register (co-executes
            // with WMMA). Lanes 0-15: K 0-7 & 16-23; lanes 16-31: K 8-15 & 24-31.
            const f32x4 q0 = *(const f32x4*)(aRowF + ko      + hi16 * 8);
            const f32x4 q1 = *(const f32x4*)(aRowF + ko      + hi16 * 8 + 4);
            const f32x4 q2 = *(const f32x4*)(aRowF + ko + 16 + hi16 * 8);
            const f32x4 q3 = *(const f32x4*)(aRowF + ko + 16 + hi16 * 8 + 4);
#pragma unroll
            for (int i = 0; i < 4; ++i) {
                a.h[0][i]     = (half_t)q0[i];
                a.h[0][4 + i] = (half_t)q1[i];
                a.h[1][i]     = (half_t)q2[i];
                a.h[1][4 + i] = (half_t)q3[i];
            }

            // B fragments (f16): lane = column o; lanes 0-15 K 0-7 & 16-23, 16-31 K 8-15 & 24-31.
            b0.h[0] = *(const v8h*)(wRow0 + ko + hi16 * 8);
            b0.h[1] = *(const v8h*)(wRow0 + ko + 16 + hi16 * 8);
            b1.h[0] = *(const v8h*)(wRow1 + ko + hi16 * 8);
            b1.h[1] = *(const v8h*)(wRow1 + ko + 16 + hi16 * 8);

            c0 = __builtin_amdgcn_wmma_f32_16x16x32_f16(false, a.v, false, b0.v,
                                                        (short)0, c0, false, false);
            c1 = __builtin_amdgcn_wmma_f32_16x16x32_f16(false, a.v, false, b1.v,
                                                        (short)0, c1, false, false);
        }

        // ---- epilogue: bias + non-temporal stores (keep x L2-resident) ------
        // C/D layout: VGPR v holds row m = v + 8*hi16, column = l16.
        const int n = n0 + s;
#pragma unroll
        for (int v = 0; v < 8; ++v) {
            const int m = v + hi16 * 8;
            const size_t row = ((size_t)(tm * 16 + m) * N_ + n) * O_;
            __builtin_nontemporal_store(c0[v] + bb0, out + row + tn0 * 16      + l16);
            __builtin_nontemporal_store(c1[v] + bb1, out + row + tn0 * 16 + 16 + l16);
        }

        // Drain this iteration's DMA, then block-wide handoff of buffers.
        asm volatile("s_wait_asynccnt 0x0" ::: "memory");
        __syncthreads();
    }
}

extern "C" void kernel_launch(void* const* d_in, const int* in_sizes, int n_in,
                              void* d_out, int out_size, void* d_ws, size_t ws_size,
                              hipStream_t stream) {
    const float* x   = (const float*)d_in[0];   // [B, N, C]
    const float* W   = (const float*)d_in[1];   // [K2, C, OUT]
    const float* b   = (const float*)d_in[2];   // [OUT]
    const int*   k2  = (const int*)  d_in[3];   // [N, K2]
    float*       out = (float*)d_out;           // [B, N, OUT]

    lattice_conv_wmma_async<<<N_ / SITES, 256, SMEM_BYTES, stream>>>(x, W, b, k2, out);
}
